// ComplexAttentionLayer_5523327943170
// MI455X (gfx1250) — matrix operand order repaired and checked by reading, not compiled, not run on hardware
//
#include <hip/hip_runtime.h>
#include <hip/hip_bf16.h>

// Problem constants
#define B_  2
#define S_  2048
#define D_  1024
#define H_  16
#define HD_ 64
#define M_  (B_ * S_)                 // 4096 tokens
#define BSD_ ((long)B_ * S_ * D_)     // 4,194,304
#define DD_  ((long)D_ * D_)          // 1,048,576
#define SCALE_ 0.125f                 // 1/sqrt(64)
#define RS16_ 2056                    // LDS row stride in bf16 (2048 + 8 pad)

typedef __bf16 bf16_t;
typedef __attribute__((ext_vector_type(16))) __bf16 v16bf;
typedef __attribute__((ext_vector_type(8)))  __bf16 bf8v;
typedef __attribute__((ext_vector_type(8)))  float  v8f;

union AFrag { bf8v h[2]; v16bf v; };

__device__ __forceinline__ v8f wmma_bf16(v16bf a, v16bf b, v8f c) {
    // D = A(16x32 bf16) * B(32x16 bf16) + C(16x16 f32)
    return __builtin_amdgcn_wmma_f32_16x16x32_bf16(
        /*neg_a=*/false, a, /*neg_b=*/false, b,
        /*c_mod=*/(short)0, c, /*reuse_a=*/false, /*reuse_b=*/false);
}

// ---------------------------------------------------------------------------
// f32 -> bf16 elementwise conversion (grid-stride)
// ---------------------------------------------------------------------------
__global__ void cvt_f32_to_bf16(const float* __restrict__ in,
                                bf16_t* __restrict__ out, long n) {
    long i = (long)blockIdx.x * blockDim.x + threadIdx.x;
    long stride = (long)gridDim.x * blockDim.x;
    for (; i < n; i += stride) out[i] = (bf16_t)in[i];
}

// ---------------------------------------------------------------------------
// Projection GEMM: C[m,n] = sum_k A[m,k] * W[n,k] + bias[n]
// A: [4096, D_] bf16 row-major; W: [D_, D_] bf16 row-major (rows = out feat)
// Each wave computes a 16x64 strip (4 WMMA accumulators, A fragment reused).
// mode 0: store q/k layout  [B,H,S,HD];  mode 1: store v transposed [B,H,HD,S]
// ---------------------------------------------------------------------------
__global__ void gemm_proj_kernel(const bf16_t* __restrict__ A,
                                 const bf16_t* __restrict__ W,
                                 const float* __restrict__ bias,
                                 bf16_t* __restrict__ out, int mode) {
    const int lane = threadIdx.x & 31;
    const int wave = threadIdx.x >> 5;
    const int sid  = blockIdx.x * 4 + wave;     // 4096 strips total
    const int MT   = M_ / 16;                   // 256
    const int tm   = sid % MT;
    const int tc   = sid / MT;                  // 0..15 (N/64)
    const int half = lane >> 4;
    const int lidx = lane & 15;

    const bf16_t* arow = A + (long)(tm * 16 + lidx) * D_;
    v8f acc[4] = {};

    for (int k0 = 0; k0 < D_; k0 += 32) {
        AFrag af;
        af.h[0] = *(const bf8v*)(arow + k0 + half * 16);
        af.h[1] = *(const bf8v*)(arow + k0 + half * 16 + 8);
#pragma unroll
        for (int j = 0; j < 4; ++j) {
            const int n = tc * 64 + j * 16 + lidx;
            v16bf bf = *(const v16bf*)(W + (long)n * D_ + k0 + half * 16);
            acc[j] = wmma_bf16(af.v, bf, acc[j]);
        }
    }

#pragma unroll
    for (int j = 0; j < 4; ++j) {
        const int ng = tc * 64 + j * 16 + lidx;   // global feature index
        const float bb = bias[ng];
        const int h = ng >> 6;                    // /HD_
        const int d = ng & (HD_ - 1);
#pragma unroll
        for (int i = 0; i < 8; ++i) {
            const int mg = tm * 16 + half * 8 + i;  // global token index
            const int b  = mg >> 11;                // /S_
            const int s  = mg & (S_ - 1);
            const float val = acc[j][i] + bb;
            long idx;
            if (mode == 0)
                idx = (((long)(b * H_ + h) * S_ + s) * HD_) + d;      // [B,H,S,HD]
            else
                idx = (((long)(b * H_ + h) * HD_ + d) * S_) + s;      // [B,H,HD,S]
            out[idx] = (bf16_t)val;
        }
    }
}

// ---------------------------------------------------------------------------
// Fused attention core: scores (WMMA) -> softmax (LDS) -> attn@v (WMMA).
// One 256-thread block (8 waves) owns a 16-row x 2048-col strip of one (b,h).
// Scores live in LDS (bf16, padded rows); attn written to d_out exactly once.
// ---------------------------------------------------------------------------
__global__ __launch_bounds__(256) void attn_fused_kernel(
        const bf16_t* __restrict__ qr, const bf16_t* __restrict__ qi,
        const bf16_t* __restrict__ kr, const bf16_t* __restrict__ ki,
        const bf16_t* __restrict__ vrt, const bf16_t* __restrict__ vit,
        const int* __restrict__ mask, float* __restrict__ attn_out,
        bf16_t* __restrict__ outr, bf16_t* __restrict__ outi) {
    __shared__ bf16_t sm[16 * RS16_];      // ~66 KB score strip
    __shared__ float  red[16][17];
    __shared__ float  rowstat[16];

    const int tid  = threadIdx.x;
    const int lane = tid & 31;
    const int wave = tid >> 5;              // 8 waves
    const int ti   = blockIdx.x & 127;      // query tile within sequence
    const int bh   = blockIdx.x >> 7;       // 0..31
    const int b    = bh / H_;
    const int h    = bh % H_;
    const int half = lane >> 4;
    const int lidx = lane & 15;

    const long qbase = (long)bh * S_ * HD_;

    // ---- Load q fragments once (2 K-steps of 32 over HD=64) ----
    const bf16_t* qrr = qr + qbase + (long)(ti * 16 + lidx) * HD_;
    const bf16_t* qir = qi + qbase + (long)(ti * 16 + lidx) * HD_;
    AFrag aqr[2], aqi[2];
#pragma unroll
    for (int kk = 0; kk < 2; ++kk) {
        const int k0 = kk * 32;
        aqr[kk].h[0] = *(const bf8v*)(qrr + k0 + half * 16);
        aqr[kk].h[1] = *(const bf8v*)(qrr + k0 + half * 16 + 8);
        aqi[kk].h[0] = *(const bf8v*)(qir + k0 + half * 16);
        aqi[kk].h[1] = *(const bf8v*)(qir + k0 + half * 16 + 8);
    }

    // ---- Phase 1: scores -> LDS (each wave does 16 of the 128 j-tiles) ----
    for (int tj = wave; tj < 128; tj += 8) {
        const bf16_t* krr = kr + qbase + (long)(tj * 16 + lidx) * HD_;
        const bf16_t* kir = ki + qbase + (long)(tj * 16 + lidx) * HD_;
        v8f acc = {};
#pragma unroll
        for (int kk = 0; kk < 2; ++kk) {
            const int k0 = kk * 32;
            v16bf bkr = *(const v16bf*)(krr + k0 + half * 16);
            v16bf bki = *(const v16bf*)(kir + k0 + half * 16);
            acc = wmma_bf16(aqr[kk].v, bkr, acc);
            acc = wmma_bf16(aqi[kk].v, bki, acc);
        }
        const int jg = tj * 16 + lidx;
        const bool masked = (mask[b * S_ + jg] == 0);
#pragma unroll
        for (int i = 0; i < 8; ++i) {
            const int r = half * 8 + i;         // row within strip
            float v = acc[i] * SCALE_;
            if (masked) v = -1e9f;
            sm[r * RS16_ + jg] = (bf16_t)v;
        }
    }
    __syncthreads();

    // ---- Phase 2: softmax over LDS strip; write attn to d_out once ----
    {
        const int r  = tid & 15;                // row
        const int t  = tid >> 4;                // chunk (128 cols each)
        bf16_t* rp = sm + r * RS16_ + t * 128;

        float mx = -3.4e38f;
        for (int c = 0; c < 128; ++c) mx = fmaxf(mx, (float)rp[c]);
        red[r][t] = mx;
        __syncthreads();
        if (t == 0) {
            float m = red[r][0];
            for (int u = 1; u < 16; ++u) m = fmaxf(m, red[r][u]);
            rowstat[r] = m;
        }
        __syncthreads();
        mx = rowstat[r];

        float sum = 0.0f;
        for (int c = 0; c < 128; ++c) {
            float e = expf((float)rp[c] - mx);
            rp[c] = (bf16_t)e;
            sum += e;
        }
        red[r][t] = sum;
        __syncthreads();
        if (t == 0) {
            float s = 0.0f;
            for (int u = 0; u < 16; ++u) s += red[r][u];
            rowstat[r] = 1.0f / s;
        }
        __syncthreads();
        const float inv = rowstat[r];

        float* gp = attn_out + ((long)bh * S_ + ti * 16 + r) * S_ + t * 128;
        for (int c = 0; c < 128; c += 4) {
            float4 vv;
            vv.x = (float)rp[c + 0] * inv;
            vv.y = (float)rp[c + 1] * inv;
            vv.z = (float)rp[c + 2] * inv;
            vv.w = (float)rp[c + 3] * inv;
            rp[c + 0] = (bf16_t)vv.x;
            rp[c + 1] = (bf16_t)vv.y;
            rp[c + 2] = (bf16_t)vv.z;
            rp[c + 3] = (bf16_t)vv.w;
            *(float4*)(gp + c) = vv;
        }
    }
    __syncthreads();

    // ---- Phase 3: attn @ v. 8 waves = (real/imag) x 4 d-tiles of HD=64 ----
    {
        const int ri = wave & 1;                 // 0 = real, 1 = imag
        const int nc = wave >> 1;                // 0..3 (16-wide d tile)
        const bf16_t* vt = ri ? vit : vrt;       // [B,H,HD,S]
        const int n = nc * 16 + lidx;            // d within head
        const bf16_t* brow = vt + ((long)bh * HD_ + n) * S_;
        const bf16_t* arow = sm + lidx * RS16_;  // attn row (bf16, normalized)

        v8f acc = {};
        for (int k0 = 0; k0 < S_; k0 += 32) {
            AFrag af;
            af.h[0] = *(const bf8v*)(arow + k0 + half * 16);
            af.h[1] = *(const bf8v*)(arow + k0 + half * 16 + 8);
            v16bf bv = *(const v16bf*)(brow + k0 + half * 16);
            acc = wmma_bf16(af.v, bv, acc);
        }

        bf16_t* outp = ri ? outi : outr;
        const int dg = h * HD_ + nc * 16 + lidx;
#pragma unroll
        for (int i = 0; i < 8; ++i) {
            const int s = ti * 16 + half * 8 + i;
            outp[((long)(b * S_ + s)) * D_ + dg] = (bf16_t)acc[i];
        }
    }
}

// ---------------------------------------------------------------------------
// Output projection GEMM: f32 result to d_out.  C = A(bf16)[4096,1024] x Wo^T + bo
// ---------------------------------------------------------------------------
__global__ void gemm_out_kernel(const bf16_t* __restrict__ A,
                                const bf16_t* __restrict__ W,
                                const float* __restrict__ bias,
                                float* __restrict__ out) {
    const int lane = threadIdx.x & 31;
    const int wave = threadIdx.x >> 5;
    const int sid  = blockIdx.x * 4 + wave;
    const int MT   = M_ / 16;
    const int tm   = sid % MT;
    const int tc   = sid / MT;
    const int half = lane >> 4;
    const int lidx = lane & 15;

    const bf16_t* arow = A + (long)(tm * 16 + lidx) * D_;
    v8f acc[4] = {};

    for (int k0 = 0; k0 < D_; k0 += 32) {
        AFrag af;
        af.h[0] = *(const bf8v*)(arow + k0 + half * 16);
        af.h[1] = *(const bf8v*)(arow + k0 + half * 16 + 8);
#pragma unroll
        for (int j = 0; j < 4; ++j) {
            const int n = tc * 64 + j * 16 + lidx;
            v16bf bf = *(const v16bf*)(W + (long)n * D_ + k0 + half * 16);
            acc[j] = wmma_bf16(af.v, bf, acc[j]);
        }
    }

#pragma unroll
    for (int j = 0; j < 4; ++j) {
        const int ng = tc * 64 + j * 16 + lidx;
        const float bb = bias[ng];
#pragma unroll
        for (int i = 0; i < 8; ++i) {
            const int mg = tm * 16 + half * 8 + i;
            out[(long)mg * D_ + ng] = acc[j][i] + bb;
        }
    }
}

// ---------------------------------------------------------------------------
extern "C" void kernel_launch(void* const* d_in, const int* in_sizes, int n_in,
                              void* d_out, int out_size, void* d_ws, size_t ws_size,
                              hipStream_t stream) {
    const float* x_real = (const float*)d_in[0];
    const float* x_imag = (const float*)d_in[1];
    const float* Wq = (const float*)d_in[2];  const float* bq = (const float*)d_in[3];
    const float* Wk = (const float*)d_in[4];  const float* bk = (const float*)d_in[5];
    const float* Wv = (const float*)d_in[6];  const float* bv = (const float*)d_in[7];
    const float* Wo = (const float*)d_in[8];  const float* bo = (const float*)d_in[9];
    const int*   mask = (const int*)d_in[10];

    float* out   = (float*)d_out;
    float* out_r = out;                  // [B,S,D]
    float* out_i = out + BSD_;           // [B,S,D]
    float* attn  = out + 2 * BSD_;       // [B,H,S,S]

    // Workspace carve-out (all 256B aligned): ~92 MB total
    char* ws = (char*)d_ws;
    size_t off = 0;
    auto alloc = [&](size_t bytes) -> void* {
        void* p = ws + off;
        off += (bytes + 255) & ~(size_t)255;
        return p;
    };
    bf16_t* xr16 = (bf16_t*)alloc(BSD_ * 2);
    bf16_t* xi16 = (bf16_t*)alloc(BSD_ * 2);
    bf16_t* wq16 = (bf16_t*)alloc(DD_ * 2);
    bf16_t* wk16 = (bf16_t*)alloc(DD_ * 2);
    bf16_t* wv16 = (bf16_t*)alloc(DD_ * 2);
    bf16_t* wo16 = (bf16_t*)alloc(DD_ * 2);
    bf16_t* qr   = (bf16_t*)alloc(BSD_ * 2);   // [B,H,S,HD]
    bf16_t* qi   = (bf16_t*)alloc(BSD_ * 2);
    bf16_t* kr   = (bf16_t*)alloc(BSD_ * 2);
    bf16_t* ki   = (bf16_t*)alloc(BSD_ * 2);
    bf16_t* vrt  = (bf16_t*)alloc(BSD_ * 2);   // [B,H,HD,S] (transposed)
    bf16_t* vit  = (bf16_t*)alloc(BSD_ * 2);
    bf16_t* or16 = (bf16_t*)alloc(BSD_ * 2);   // [B,S,D]
    bf16_t* oi16 = (bf16_t*)alloc(BSD_ * 2);
    (void)ws_size; (void)in_sizes; (void)n_in; (void)out_size;

    // 1) bf16 conversions
    cvt_f32_to_bf16<<<2048, 256, 0, stream>>>(x_real, xr16, BSD_);
    cvt_f32_to_bf16<<<2048, 256, 0, stream>>>(x_imag, xi16, BSD_);
    cvt_f32_to_bf16<<<1024, 256, 0, stream>>>(Wq, wq16, DD_);
    cvt_f32_to_bf16<<<1024, 256, 0, stream>>>(Wk, wk16, DD_);
    cvt_f32_to_bf16<<<1024, 256, 0, stream>>>(Wv, wv16, DD_);
    cvt_f32_to_bf16<<<1024, 256, 0, stream>>>(Wo, wo16, DD_);

    // 2) q/k/v projections (WMMA)
    gemm_proj_kernel<<<1024, 128, 0, stream>>>(xr16, wq16, bq, qr, 0);
    gemm_proj_kernel<<<1024, 128, 0, stream>>>(xi16, wq16, bq, qi, 0);
    gemm_proj_kernel<<<1024, 128, 0, stream>>>(xr16, wk16, bk, kr, 0);
    gemm_proj_kernel<<<1024, 128, 0, stream>>>(xi16, wk16, bk, ki, 0);
    gemm_proj_kernel<<<1024, 128, 0, stream>>>(xr16, wv16, bv, vrt, 1);
    gemm_proj_kernel<<<1024, 128, 0, stream>>>(xi16, wv16, bv, vit, 1);

    // 3) fused scores + softmax + attn@v  (attn hits HBM exactly once)
    attn_fused_kernel<<<4096, 256, 0, stream>>>(qr, qi, kr, ki, vrt, vit,
                                                mask, attn, or16, oi16);

    // 4) output projections (WMMA) -> f32 d_out
    gemm_out_kernel<<<1024, 128, 0, stream>>>(or16, wo16, bo, out_r);
    gemm_out_kernel<<<1024, 128, 0, stream>>>(oi16, wo16, bo, out_i);
}
